// LBLResNetBiLm_71021579206698
// MI455X (gfx1250) — matrix operand order repaired and checked by reading, not compiled, not run on hardware
//
#include <hip/hip_runtime.h>
#include <hip/hip_bf16.h>

// Problem constants (reference: B=16, S=2048, D=256, W=8, L=3)
#define BB    16
#define SS    2048
#define DD    256
#define WW    8
#define LL    3
#define MROWS 64                 // rows of x per workgroup
#define BSROWS (BB * SS)         // 32768 total rows

typedef __attribute__((ext_vector_type(16))) _Float16     v16h;
typedef __attribute__((ext_vector_type(8)))  float        v8f;
typedef __attribute__((ext_vector_type(4)))  unsigned int v4u;

union Frag16 { v16h v; v4u q[2]; };   // 32 bytes: one WMMA f16 A/B fragment per lane

static __device__ __forceinline__ v8f zero8() {
    v8f z = {0.f, 0.f, 0.f, 0.f, 0.f, 0.f, 0.f, 0.f};
    return z;
}

// ---------------------------------------------------------------------------
// Stage 0: convert the 12 D x D fp32 weight matrices to f16 in workspace.
// Layout: [branch(2)][layer(3)][w1/w2(2)] contiguous 65536-half chunks.
// ---------------------------------------------------------------------------
__global__ __launch_bounds__(256) void cvt_weights_f16(
    const float* __restrict__ lw1, const float* __restrict__ lw2,
    const float* __restrict__ rw1, const float* __restrict__ rw2,
    _Float16* __restrict__ dst)
{
    int i = blockIdx.x * 256 + threadIdx.x;       // 0 .. 12*65536-1
    int mat = i >> 16;                            // 0..11  == c*6 + layer*2 + which
    int off = i & 65535;
    int c     = mat / 6;
    int rem   = mat % 6;
    int layer = rem >> 1;
    int which = rem & 1;
    const float* src = c ? (which ? rw2 : rw1) : (which ? lw2 : lw1);
    dst[i] = (_Float16)src[layer * 65536 + off];
}

// ---------------------------------------------------------------------------
// One wave computes a 16(M) x 128(N) tile of  A(64x256,f16 LDS) @ B(256x256,f16)
// with v_wmma_f32_16x16x32_f16, accumulating into acc[8] (8 N-tiles).
//
// A layout (16-bit A 16x32, ISA 7.12.2): lane = M (mod 16); halves 0-7 hold
// K in {0..7}+8*lane[4], halves 8-15 hold K in {16..23}+8*lane[4]  -> two
// contiguous 16-byte LDS loads per K-step of 32.
// B layout (16-bit B, ISA 7.12.4 pattern): lane = N (mod 16), halves = K (same order as A)
// -> 16 strided 2-byte global loads per N-tile.
// ---------------------------------------------------------------------------
static __device__ __forceinline__ void wave_gemm_64x256(
    const _Float16* __restrict__ A,     // LDS, row-major 64 x 256
    const _Float16* __restrict__ Bw,    // global f16, row-major 256 x 256 (K x N)
    v8f acc[8], int mtile, int nbase, int lane)
{
    const int mrow = mtile * 16 + (lane & 15);
    #pragma unroll
    for (int kt = 0; kt < 8; ++kt) {
        Frag16 a;
        const int kb = kt * 32 + ((lane & 16) >> 1);         // K group base (0 or 8)
        a.q[0] = *(const v4u*)(A + mrow * DD + kb);          // K = kb .. kb+7
        a.q[1] = *(const v4u*)(A + mrow * DD + kb + 16);     // K = kb+16 .. kb+23
        const int krow = kt * 32 + ((lane & 16) >> 1);       // B: lane indexes N, halves K (same order as A)
        const _Float16* bp = Bw + krow * DD + nbase + (lane & 15);
        #pragma unroll
        for (int nt = 0; nt < 8; ++nt) {
            Frag16 b;
            #pragma unroll
            for (int i = 0; i < 8; ++i) {
                b.v[i]     = bp[i * DD + nt * 16];
                b.v[8 + i] = bp[(16 + i) * DD + nt * 16];
            }
            acc[nt] = __builtin_amdgcn_wmma_f32_16x16x32_f16(
                false, a.v, false, b.v, (short)0, acc[nt], false, false);
        }
    }
}

// ---------------------------------------------------------------------------
// Fused kernel: conv (9-tap window) -> 3 x (LN -> GEMM1+ReLU -> GEMM2+residual)
// for one branch (blockIdx.y) and one 64-row block (blockIdx.x).
// LDS: x (64x256 f32) | xhat (64x256 f16) | h (64x256 f16)  = 128 KB.
// ---------------------------------------------------------------------------
__global__ __launch_bounds__(256) void lbl_fused_kernel(
    const float* __restrict__ inp,
    const float* __restrict__ lpad, const float* __restrict__ rpad,
    const float* __restrict__ lwts, const float* __restrict__ rwts,
    const float* __restrict__ lb1,  const float* __restrict__ lb2,
    const float* __restrict__ lg,   const float* __restrict__ lbe,
    const float* __restrict__ rb1,  const float* __restrict__ rb2,
    const float* __restrict__ rg,   const float* __restrict__ rbe,
    const _Float16* __restrict__ wh,     // 12 * 65536 f16 weights
    float* __restrict__ out)
{
    extern __shared__ char smem[];
    float*    x  = (float*)smem;                         // 64 KB
    _Float16* xh = (_Float16*)(smem + 64 * 1024);        // 32 KB
    _Float16* hb = (_Float16*)(smem + 96 * 1024);        // 32 KB

    const int c    = blockIdx.y;                 // 0 = left, 1 = right
    const int r0   = blockIdx.x * MROWS;         // global row base (b*S + s)
    const int b    = r0 >> 11;                   // S = 2048
    const int s0   = r0 & (SS - 1);
    const int tid  = threadIdx.x;
    const int lane = tid & 31;
    const int wv   = tid >> 5;                   // wave id, 0..7 (wave32)

    // ---- Stage 1: 9-tap windowed sum over padded sequence -> x ----
    const float* wts = c ? rwts : lwts;
    float wreg[WW + 1];
    #pragma unroll
    for (int w = 0; w <= WW; ++w) wreg[w] = wts[w];
    const int joff = c ? WW : 0;                 // right branch windows shift by W

    for (int idx = tid; idx < MROWS * DD; idx += 256) {
        const int m = idx >> 8, d = idx & 255;
        const int s = s0 + m;
        float acc = 0.f;
        #pragma unroll
        for (int w = 0; w <= WW; ++w) {
            const int j = s + w + joff;          // index into padded rows [0, S+2W)
            float v;
            if (j < WW)            v = lpad[j * DD + d];
            else if (j < WW + SS)  v = inp[((size_t)b * SS + (j - WW)) * DD + d];
            else                   v = rpad[(j - WW - SS) * DD + d];
            acc += wreg[w] * v;
        }
        x[idx] = acc;
    }
    __syncthreads();

    const float* g_all  = c ? rg  : lg;
    const float* be_all = c ? rbe : lbe;
    const float* b1_all = c ? rb1 : lb1;
    const float* b2_all = c ? rb2 : lb2;

    const int mtile = wv & 3;                    // 4 M-tiles of 16 rows
    const int nbase = (wv >> 2) * 128;           // 2 N-halves of 128 cols
    const int hi    = lane >> 4;                 // C-layout: M = v + 8*hi
    const int ncol  = lane & 15;                 // C-layout: N = ncol (+tile base)

    for (int layer = 0; layer < LL; ++layer) {
        const float* g  = g_all  + layer * DD;
        const float* be = be_all + layer * DD;
        const float* b1 = b1_all + layer * DD;
        const float* b2 = b2_all + layer * DD;
        const _Float16* w1h = wh + (size_t)(c * 6 + layer * 2 + 0) * 65536;
        const _Float16* w2h = wh + (size_t)(c * 6 + layer * 2 + 1) * 65536;

        // ---- LayerNorm: wave wv owns rows wv*8 .. wv*8+7; lane owns 8 cols ----
        #pragma unroll
        for (int rr = 0; rr < 8; ++rr) {
            const int m = wv * 8 + rr;
            float vals[8], sum = 0.f, ss = 0.f;
            #pragma unroll
            for (int j = 0; j < 8; ++j) {
                vals[j] = x[m * DD + lane * 8 + j];
                sum += vals[j];
                ss  += vals[j] * vals[j];
            }
            #pragma unroll
            for (int off = 16; off > 0; off >>= 1) {   // wave32 butterfly reduce
                sum += __shfl_xor(sum, off, 32);
                ss  += __shfl_xor(ss,  off, 32);
            }
            const float mean = sum * (1.f / DD);
            const float var  = ss  * (1.f / DD) - mean * mean;
            const float rstd = rsqrtf(var + 1e-6f);
            #pragma unroll
            for (int j = 0; j < 8; ++j) {
                const int d = lane * 8 + j;
                xh[m * DD + d] = (_Float16)(g[d] * (vals[j] - mean) * rstd + be[d]);
            }
        }
        __syncthreads();

        // ---- GEMM1: h = relu(xhat @ w1 + b1) ----
        v8f acc[8];
        #pragma unroll
        for (int nt = 0; nt < 8; ++nt) acc[nt] = zero8();
        wave_gemm_64x256(xh, w1h, acc, mtile, nbase, lane);

        #pragma unroll
        for (int nt = 0; nt < 8; ++nt) {
            const int n = nbase + nt * 16 + ncol;
            const float bias = b1[n];
            #pragma unroll
            for (int v = 0; v < 8; ++v) {
                const int m = mtile * 16 + v + 8 * hi;
                hb[m * DD + n] = (_Float16)fmaxf(acc[nt][v] + bias, 0.f);
            }
        }
        __syncthreads();

        // ---- GEMM2: y = h @ w2 + b2 ; x += y ; emit layer slice ----
        #pragma unroll
        for (int nt = 0; nt < 8; ++nt) acc[nt] = zero8();
        wave_gemm_64x256(hb, w2h, acc, mtile, nbase, lane);

        #pragma unroll
        for (int nt = 0; nt < 8; ++nt) {
            const int n = nbase + nt * 16 + ncol;
            const float bias = b2[n];
            #pragma unroll
            for (int v = 0; v < 8; ++v) {
                const int m  = mtile * 16 + v + 8 * hi;
                const float xv = x[m * DD + n] + acc[nt][v] + bias;
                x[m * DD + n] = xv;
                const size_t r = (size_t)(r0 + m);
                // all_layers: (L, B, S, 2D)
                out[((size_t)layer * BSROWS + r) * 512 + (size_t)c * 256 + n] = xv;
                if (layer == LL - 1)   // duplicate final layer: (B, S, 2D) tail
                    out[((size_t)LL * BSROWS + r) * 512 + (size_t)c * 256 + n] = xv;
            }
        }
        __syncthreads();
    }
}

// ---------------------------------------------------------------------------
extern "C" void kernel_launch(void* const* d_in, const int* in_sizes, int n_in,
                              void* d_out, int out_size, void* d_ws, size_t ws_size,
                              hipStream_t stream)
{
    (void)in_sizes; (void)n_in; (void)out_size; (void)ws_size;
    const float* inp  = (const float*)d_in[0];
    const float* lpad = (const float*)d_in[1];
    const float* rpad = (const float*)d_in[2];
    const float* lwts = (const float*)d_in[3];
    const float* rwts = (const float*)d_in[4];
    const float* lw1  = (const float*)d_in[5];
    const float* lb1  = (const float*)d_in[6];
    const float* lw2  = (const float*)d_in[7];
    const float* lb2  = (const float*)d_in[8];
    const float* lg   = (const float*)d_in[9];
    const float* lbe  = (const float*)d_in[10];
    const float* rw1  = (const float*)d_in[11];
    const float* rb1  = (const float*)d_in[12];
    const float* rw2  = (const float*)d_in[13];
    const float* rb2  = (const float*)d_in[14];
    const float* rg   = (const float*)d_in[15];
    const float* rbe  = (const float*)d_in[16];

    _Float16* wh = (_Float16*)d_ws;              // 12 * 65536 halves = 1.5 MB

    // Stage 0: fp32 -> f16 weight conversion (weights then live in L2).
    cvt_weights_f16<<<(12 * 65536) / 256, 256, 0, stream>>>(lw1, lw2, rw1, rw2, wh);

    // Allow 128 KB dynamic LDS (WGP supports up to 320 KB).
    static_assert(MROWS * DD * 4 + 2 * MROWS * DD * 2 == 131072, "lds size");
    hipFuncSetAttribute((const void*)lbl_fused_kernel,
                        hipFuncAttributeMaxDynamicSharedMemorySize, 131072);

    dim3 grid(BSROWS / MROWS, 2);                // 512 row-blocks x 2 branches
    lbl_fused_kernel<<<grid, 256, 131072, stream>>>(
        inp, lpad, rpad, lwts, rwts,
        lb1, lb2, lg, lbe,
        rb1, rb2, rg, rbe,
        wh, (float*)d_out);
}
